// ThinkingRNN_7524782702942
// MI455X (gfx1250) — matrix operand order repaired and checked by reading, not compiled, not run on hardware
//
#include <hip/hip_runtime.h>
#include <hip/hip_bf16.h>
#include <stdint.h>
#include <stddef.h>

// ---------------- problem constants ----------------
#define Bb 128
#define Tt 515
#define Vv 512
#define Hh 1024
#define NDEC 512          // T - 3
#define ENDTOK (Vv - 1)
#define THINK_STEPS 33    // MAX_THINK_STEPS + 1

typedef _Float16 half_t;
typedef __attribute__((ext_vector_type(16))) _Float16 v16h;
typedef __attribute__((ext_vector_type(8)))  _Float16 v8h;
typedef __attribute__((ext_vector_type(8)))  float    v8f;

// ---------------- WMMA fragment loaders (wave32, 16x16x32 f16) ----------------
// A fragment (16x32, f16 source). Lane l<16: M=l, K = k0..k0+7 and k0+16..k0+23.
// Lane l>=16: M=l-16, K = k0+8..k0+15 and k0+24..k0+31. (ISA 7.12.2)
__device__ inline v16h load_a_f16(const half_t* __restrict__ A, int ldA,
                                  int m0, int k0, int lane) {
    int m  = m0 + (lane & 15);
    int kb = k0 + ((lane & 16) ? 8 : 0);
    const half_t* p = A + (size_t)m * ldA + kb;
    v8h lo = *(const v8h*)(p);
    v8h hi = *(const v8h*)(p + 16);
    v16h r;
#pragma unroll
    for (int i = 0; i < 8; ++i) { r[i] = lo[i]; r[i + 8] = hi[i]; }
    return r;
}

// Same fragment but sourced from f32 memory (converted in registers).
__device__ inline v16h load_a_f32(const float* __restrict__ A, size_t rowStride,
                                  size_t base, int m0, int k0, int lane) {
    int m  = m0 + (lane & 15);
    int kb = k0 + ((lane & 16) ? 8 : 0);
    const float* p = A + (size_t)m * rowStride + base + kb;
    v8f lo = *(const v8f*)(p);
    v8f hi = *(const v8f*)(p + 16);
    v16h r;
#pragma unroll
    for (int i = 0; i < 8; ++i) {
        r[i]     = (_Float16)lo[i];
        r[i + 8] = (_Float16)hi[i];
    }
    return r;
}

// B fragment (32x16): B[k][n] = W[n][k], W row-major [N][K].
// Lane l<16: n=l, K=k0..k0+7 and k0+16..k0+23. Lane l>=16: n=l-16, K=k0+8..k0+15 and k0+24..k0+31.
__device__ inline v16h load_b_frag(const half_t* __restrict__ W, int ldW,
                                   int n0, int k0, int lane) {
    int n  = n0 + (lane & 15);
    int kb = k0 + ((lane & 16) ? 8 : 0);
    const half_t* p = W + (size_t)n * ldW + kb;
    v8h lo = *(const v8h*)(p);
    v8h hi = *(const v8h*)(p + 16);
    v16h r;
#pragma unroll
    for (int i = 0; i < 8; ++i) { r[i] = lo[i]; r[i + 8] = hi[i]; }
    return r;
}

// ---------------- prep: f32->f16 weights, fused bias, zero state ----------------
__global__ void prep_kernel(const float* __restrict__ W_ih, const float* __restrict__ W_hh,
                            const float* __restrict__ b_ih, const float* __restrict__ b_hh,
                            const float* __restrict__ W_fc,
                            half_t* __restrict__ W_ih16, half_t* __restrict__ W_hh16,
                            half_t* __restrict__ W_fc16, float* __restrict__ bias_sum,
                            half_t* __restrict__ h0, half_t* __restrict__ h1,
                            int* __restrict__ done) {
    size_t i      = (size_t)blockIdx.x * blockDim.x + threadIdx.x;
    size_t stride = (size_t)gridDim.x * blockDim.x;
    for (size_t k = i; k < (size_t)Hh * Vv; k += stride) W_ih16[k] = (half_t)W_ih[k];
    for (size_t k = i; k < (size_t)Hh * Hh; k += stride) W_hh16[k] = (half_t)W_hh[k];
    for (size_t k = i; k < (size_t)Vv * Hh; k += stride) W_fc16[k] = (half_t)W_fc[k];
    for (size_t k = i; k < Hh; k += stride) bias_sum[k] = b_ih[k] + b_hh[k];
    for (size_t k = i; k < (size_t)Bb * Hh; k += stride) { h0[k] = (half_t)0.f; h1[k] = (half_t)0.f; }
    for (size_t k = i; k < Bb; k += stride) done[k] = 0;
}

// ---------------- RNN cell: h_out = tanh(inp@W_ih^T + h@W_hh^T + bias) ----------------
// Grid: (M/64, N/128). Block: 256 threads = 8 waves; waves tiled 4(row) x 2(col);
// each wave owns a 16x64 strip = 4 wmma accumulators. useX branch hoisted out of
// the K-loop; K-loops unrolled x2 to deepen the load pipeline over the WMMA chain.
__global__ __launch_bounds__(256)
void cell_kernel(const float* __restrict__ x, int t, int useX,
                 const half_t* __restrict__ curr,
                 const half_t* __restrict__ h_in,
                 const half_t* __restrict__ W_ih16,
                 const half_t* __restrict__ W_hh16,
                 const float* __restrict__ bias,
                 half_t* __restrict__ h_out) {
    int lane = threadIdx.x & 31;
    int wave = threadIdx.x >> 5;
    int wrow = wave >> 1, wcol = wave & 1;
    int m0  = blockIdx.x * 64 + wrow * 16;
    int nw0 = blockIdx.y * 128 + wcol * 64;

    v8f acc[4];
#pragma unroll
    for (int j = 0; j < 4; ++j)
#pragma unroll
        for (int r = 0; r < 8; ++r) acc[j][r] = 0.0f;

    // K segment 1: input term (K = V). Uniform branch hoisted out of the loop.
    if (useX) {
#pragma unroll 2
        for (int k0 = 0; k0 < Vv; k0 += 32) {
            v16h a = load_a_f32(x, (size_t)Tt * Vv, (size_t)t * Vv, m0, k0, lane);
#pragma unroll
            for (int j = 0; j < 4; ++j) {
                v16h bf = load_b_frag(W_ih16, Vv, nw0 + j * 16, k0, lane);
                acc[j] = __builtin_amdgcn_wmma_f32_16x16x32_f16(
                    false, a, false, bf, (short)0, acc[j], false, false);
            }
        }
    } else {
#pragma unroll 2
        for (int k0 = 0; k0 < Vv; k0 += 32) {
            v16h a = load_a_f16(curr, Vv, m0, k0, lane);
#pragma unroll
            for (int j = 0; j < 4; ++j) {
                v16h bf = load_b_frag(W_ih16, Vv, nw0 + j * 16, k0, lane);
                acc[j] = __builtin_amdgcn_wmma_f32_16x16x32_f16(
                    false, a, false, bf, (short)0, acc[j], false, false);
            }
        }
    }
    // K segment 2: hidden term (K = H).
#pragma unroll 2
    for (int k0 = 0; k0 < Hh; k0 += 32) {
        v16h a = load_a_f16(h_in, Hh, m0, k0, lane);
#pragma unroll
        for (int j = 0; j < 4; ++j) {
            v16h bf = load_b_frag(W_hh16, Hh, nw0 + j * 16, k0, lane);
            acc[j] = __builtin_amdgcn_wmma_f32_16x16x32_f16(
                false, a, false, bf, (short)0, acc[j], false, false);
        }
    }
    // Epilogue: bias + tanh, store f16. C layout: VGPR r -> row m0+r (lanes 0-15)
    // or m0+8+r (lanes 16-31); col = n0 + (lane&15).
    int col_lo = lane & 15;
    int rbase  = m0 + ((lane & 16) ? 8 : 0);
#pragma unroll
    for (int j = 0; j < 4; ++j) {
        int col  = nw0 + j * 16 + col_lo;
        float bv = bias[col];
#pragma unroll
        for (int r = 0; r < 8; ++r) {
            float v = tanhf(acc[j][r] + bv);
            h_out[(size_t)(rbase + r) * Hh + col] = (half_t)v;
        }
    }
}

// ---------------- logits = h @ W_fc^T + b_fc (f32 out) ----------------
__global__ __launch_bounds__(256)
void fc_kernel(const half_t* __restrict__ h,
               const half_t* __restrict__ W_fc16,
               const float* __restrict__ b_fc,
               float* __restrict__ logits) {
    int lane = threadIdx.x & 31;
    int wave = threadIdx.x >> 5;
    int wrow = wave >> 1, wcol = wave & 1;
    int m0  = blockIdx.x * 64 + wrow * 16;
    int nw0 = blockIdx.y * 128 + wcol * 64;

    v8f acc[4];
#pragma unroll
    for (int j = 0; j < 4; ++j)
#pragma unroll
        for (int r = 0; r < 8; ++r) acc[j][r] = 0.0f;

#pragma unroll 2
    for (int k0 = 0; k0 < Hh; k0 += 32) {
        v16h a = load_a_f16(h, Hh, m0, k0, lane);
#pragma unroll
        for (int j = 0; j < 4; ++j) {
            v16h bf = load_b_frag(W_fc16, Hh, nw0 + j * 16, k0, lane);
            acc[j] = __builtin_amdgcn_wmma_f32_16x16x32_f16(
                false, a, false, bf, (short)0, acc[j], false, false);
        }
    }
    int col_lo = lane & 15;
    int rbase  = m0 + ((lane & 16) ? 8 : 0);
#pragma unroll
    for (int j = 0; j < 4; ++j) {
        int col  = nw0 + j * 16 + col_lo;
        float bv = b_fc[col];
#pragma unroll
        for (int r = 0; r < 8; ++r)
            logits[(size_t)(rbase + r) * Vv + col] = acc[j][r] + bv;
    }
}

// ---------------- gumbel-argmax + state commit ----------------
__device__ inline float gumbel_noise(unsigned seed, int step, int b, int v) {
    uint64_t z = ((uint64_t)seed << 32)
               ^ ((uint64_t)(unsigned)step * 0x9E3779B97F4A7C15ull)
               ^ ((uint64_t)(unsigned)b    * 0xBF58476D1CE4E5B9ull)
               ^ ((uint64_t)(unsigned)v    * 0x94D049BB133111EBull);
    z += 0x9E3779B97F4A7C15ull;
    z = (z ^ (z >> 30)) * 0xBF58476D1CE4E5B9ull;
    z = (z ^ (z >> 27)) * 0x94D049BB133111EBull;
    z ^= (z >> 31);
    float u = (float)((z >> 40) + 1) * (1.0f / 16777218.0f);  // in (0,1)
    return -logf(-logf(u));
}

// phase 0 = think (done-masked), phase 1 = decode (writes one-hot output row)
__global__ __launch_bounds__(256)
void update_kernel(const float* __restrict__ logits,
                   const half_t* __restrict__ h_new, half_t* __restrict__ h_cur,
                   half_t* __restrict__ curr, int* __restrict__ done,
                   float* __restrict__ out, int phase, int step, unsigned seed) {
    __shared__ float sval[256];
    __shared__ int   sidx[256];
    __shared__ int   widx;
    int b = blockIdx.x, tid = threadIdx.x;

    float best = -3.4e38f; int bi = 0;
    for (int v = tid; v < Vv; v += 256) {
        float val = logits[(size_t)b * Vv + v] + gumbel_noise(seed, step, b, v);
        if (val > best) { best = val; bi = v; }
    }
    sval[tid] = best; sidx[tid] = bi;
    __syncthreads();
    for (int s = 128; s > 0; s >>= 1) {
        if (tid < s) {
            bool take = (sval[tid + s] > sval[tid]) ||
                        (sval[tid + s] == sval[tid] && sidx[tid + s] < sidx[tid]);
            if (take) { sval[tid] = sval[tid + s]; sidx[tid] = sidx[tid + s]; }
        }
        __syncthreads();
    }
    if (tid == 0) widx = sidx[0];
    __syncthreads();
    int idx = widx;
    int old_done = done[b];

    if (phase == 0) {  // think: commit only if not done (uses pre-update done)
        if (!old_done) {
            for (int i = tid; i < Hh; i += 256)
                h_cur[(size_t)b * Hh + i] = h_new[(size_t)b * Hh + i];
            for (int v = tid; v < Vv; v += 256)
                curr[(size_t)b * Vv + v] = (half_t)((v == idx) ? 1.0f : 0.0f);
            if (tid == 0 && idx == ENDTOK) done[b] = 1;
        }
    } else {           // decode: unconditional commit, emit one-hot (vocab sliced to V-1)
        for (int i = tid; i < Hh; i += 256)
            h_cur[(size_t)b * Hh + i] = h_new[(size_t)b * Hh + i];
        for (int v = tid; v < Vv; v += 256)
            curr[(size_t)b * Vv + v] = (half_t)((v == idx) ? 1.0f : 0.0f);
        if (tid == 0 && idx < ENDTOK)
            out[((size_t)b * NDEC + step) * (Vv - 1) + idx] = 1.0f;
    }
}

// ---------------- host orchestration ----------------
extern "C" void kernel_launch(void* const* d_in, const int* in_sizes, int n_in,
                              void* d_out, int out_size, void* d_ws, size_t ws_size,
                              hipStream_t stream) {
    const float* x    = (const float*)d_in[0];
    const float* W_ih = (const float*)d_in[1];
    const float* W_hh = (const float*)d_in[2];
    const float* b_ih = (const float*)d_in[3];
    const float* b_hh = (const float*)d_in[4];
    const float* W_fc = (const float*)d_in[5];
    const float* b_fc = (const float*)d_in[6];

    char* ws = (char*)d_ws;
    size_t off = 0;
    auto alloc = [&](size_t bytes) { char* p = ws + off; off = (off + bytes + 255) & ~(size_t)255; return p; };
    half_t* W_ih16 = (half_t*)alloc((size_t)Hh * Vv * 2);
    half_t* W_hh16 = (half_t*)alloc((size_t)Hh * Hh * 2);
    half_t* W_fc16 = (half_t*)alloc((size_t)Vv * Hh * 2);
    float*  biasS  = (float*) alloc((size_t)Hh * 4);
    half_t* h0     = (half_t*)alloc((size_t)Bb * Hh * 2);
    half_t* h1     = (half_t*)alloc((size_t)Bb * Hh * 2);
    half_t* curr16 = (half_t*)alloc((size_t)Bb * Vv * 2);
    float*  logits = (float*) alloc((size_t)Bb * Vv * 4);
    int*    done   = (int*)   alloc((size_t)Bb * 4);

    hipMemsetAsync(d_out, 0, (size_t)out_size * sizeof(float), stream);
    prep_kernel<<<1024, 256, 0, stream>>>(W_ih, W_hh, b_ih, b_hh, W_fc,
                                          W_ih16, W_hh16, W_fc16, biasS, h0, h1, done);

    half_t* hbuf[2] = {h0, h1};
    dim3 cellGrid(Bb / 64, Hh / 128);   // (2, 8)
    dim3 fcGrid(Bb / 64, Vv / 128);     // (2, 4)

    // Encode: 515 steps, h ping-pong via host pointer swap.
    int cur = 0;
    for (int t = 0; t < Tt; ++t) {
        cell_kernel<<<cellGrid, 256, 0, stream>>>(x, t, /*useX=*/1, curr16,
                                                  hbuf[cur], W_ih16, W_hh16, biasS,
                                                  hbuf[cur ^ 1]);
        cur ^= 1;
    }
    half_t* h_cur = hbuf[cur];
    half_t* h_new = hbuf[cur ^ 1];

    // Think: 33 steps (step 0 uses curr = x[:, T-1, :], then one-hot curr16).
    for (int s = 0; s < THINK_STEPS; ++s) {
        cell_kernel<<<cellGrid, 256, 0, stream>>>(x, Tt - 1, /*useX=*/(s == 0) ? 1 : 0,
                                                  curr16, h_cur, W_ih16, W_hh16, biasS, h_new);
        fc_kernel<<<fcGrid, 256, 0, stream>>>(h_new, W_fc16, b_fc, logits);
        update_kernel<<<Bb, 256, 0, stream>>>(logits, h_new, h_cur, curr16, done,
                                              (float*)d_out, /*phase=*/0, s, 42u);
    }
    // Decode: 512 steps, emit one-hot rows.
    for (int s = 0; s < NDEC; ++s) {
        cell_kernel<<<cellGrid, 256, 0, stream>>>(x, 0, /*useX=*/0, curr16,
                                                  h_cur, W_ih16, W_hh16, biasS, h_new);
        fc_kernel<<<fcGrid, 256, 0, stream>>>(h_new, W_fc16, b_fc, logits);
        update_kernel<<<Bb, 256, 0, stream>>>(logits, h_new, h_cur, curr16, done,
                                              (float*)d_out, /*phase=*/1, s, 7u);
    }
}
